// GCCL_82686710383006
// MI455X (gfx1250) — matrix operand order repaired and checked by reading, not compiled, not run on hardware
//
#include <hip/hip_runtime.h>

// ---------------------------------------------------------------------------
// GCCL for MI455X (gfx1250): precomputed bf16 a_norm + TDM (tensor_load_to_lds)
// double-buffered bf16 WMMA GEMMs + fused contrastive loss.
// ---------------------------------------------------------------------------

typedef __bf16 bf16_t;
typedef bf16_t v16bf __attribute__((ext_vector_type(16)));
typedef float  v8f   __attribute__((ext_vector_type(8)));
typedef unsigned int   uint32;
typedef unsigned short ush;
typedef unsigned int  u32x4 __attribute__((ext_vector_type(4)));
typedef int           i32x8 __attribute__((ext_vector_type(8)));
typedef int           i32x4 __attribute__((ext_vector_type(4)));

#define GN 6144
#define GD 2000
#define GH 512
#define GP 256

#if __has_builtin(__builtin_amdgcn_tensor_load_to_lds) && __has_builtin(__builtin_amdgcn_s_wait_tensorcnt)
#define HAVE_TDM 1
#else
#define HAVE_TDM 0
#endif

union Frag {
    v16bf v;
    uint4 q[2];
    uint32 u[8];
};

__device__ __forceinline__ ush f2bf(float f) {
    uint32 u = __float_as_uint(f);
    uint32 r = u + 0x7FFFu + ((u >> 16) & 1u);   // RNE
    return (ush)(r >> 16);
}

#if __has_builtin(__builtin_amdgcn_cvt_pk_bf16_f32)
typedef bf16_t v2bf __attribute__((ext_vector_type(2)));
__device__ __forceinline__ uint32 pack2bf(float a, float b) {
    union { v2bf v; uint32 u; } cv;
    cv.v = __builtin_amdgcn_cvt_pk_bf16_f32(a, b);
    return cv.u;
}
#else
__device__ __forceinline__ uint32 rne_u(float f) {
    uint32 u = __float_as_uint(f);
    return u + 0x7FFFu + ((u >> 16) & 1u);
}
__device__ __forceinline__ uint32 pack2bf(float a, float b) {
    // D.b16lo = a[31:16], D.b16hi = b[31:16]  (one v_perm_b32)
    return __builtin_amdgcn_perm(rne_u(b), rne_u(a), 0x07060302u);
}
#endif

__device__ __forceinline__ uint4 pack8bf(float4 a, float4 b) {
    uint4 q;
    q.x = pack2bf(a.x, a.y); q.y = pack2bf(a.z, a.w);
    q.z = pack2bf(b.x, b.y); q.w = pack2bf(b.z, b.w);
    return q;
}

// ---------------------------------------------------------------------------
// Tiling
// ---------------------------------------------------------------------------
#define BM 128
#define BN 64
#define BK 32
#define SA 40   // LDS halfword stride: 32 data + 8 pad (16B-aligned fragments)
#define SB 40
#define ASZ (BM * SA)
#define BSZ (BN * SB)

// ---------------------------------------------------------------------------
// a_norm precompute: out[i][j] = bf16((adj(*emask) + I) * dis[i]*dis[j])
// ---------------------------------------------------------------------------
template<bool USEE>
__global__ __launch_bounds__(256)
void prep_anorm(const float* __restrict__ adj, const float* __restrict__ emask,
                const float* __restrict__ dis, ush* __restrict__ out)
{
    const int row = blockIdx.x, tid = threadIdx.x;
    const float di_ = dis[row];
    const float* ap = adj + (size_t)row * GN;
    const float* mp = emask + (size_t)row * GN;
    ush* op = out + (size_t)row * GN;
    for (int c = tid; c < GN / 8; c += 256) {
        const int j = c * 8;
        float4 a0 = *(const float4*)(ap + j);
        float4 a1 = *(const float4*)(ap + j + 4);
        if (USEE) {
            float4 m0v = *(const float4*)(mp + j);
            float4 m1v = *(const float4*)(mp + j + 4);
            a0.x *= m0v.x; a0.y *= m0v.y; a0.z *= m0v.z; a0.w *= m0v.w;
            a1.x *= m1v.x; a1.y *= m1v.y; a1.z *= m1v.z; a1.w *= m1v.w;
        }
        const int d = row - j;                           // identity (branchless)
        a0.x += (d == 0) ? 1.f : 0.f; a0.y += (d == 1) ? 1.f : 0.f;
        a0.z += (d == 2) ? 1.f : 0.f; a0.w += (d == 3) ? 1.f : 0.f;
        a1.x += (d == 4) ? 1.f : 0.f; a1.y += (d == 5) ? 1.f : 0.f;
        a1.z += (d == 6) ? 1.f : 0.f; a1.w += (d == 7) ? 1.f : 0.f;
        const float4 d0 = *(const float4*)(dis + j);
        const float4 d1 = *(const float4*)(dis + j + 4);
        a0.x *= di_ * d0.x; a0.y *= di_ * d0.y; a0.z *= di_ * d0.z; a0.w *= di_ * d0.w;
        a1.x *= di_ * d1.x; a1.y *= di_ * d1.y; a1.z *= di_ * d1.z; a1.w *= di_ * d1.w;
        *(uint4*)(op + j) = pack8bf(a0, a1);
    }
}

// ---------------------------------------------------------------------------
// fp32 B-tile staging (transposed into LDS as bf16)
// ---------------------------------------------------------------------------
__device__ __forceinline__ void stage_b_f32(
    ush* __restrict__ dst, const float* __restrict__ B,
    int n0, int k0, int K, int Nb, int tid)
{
    const int kb = tid >> 3;          // 0..31
    const int ns = (tid & 7) * 8;     // 0..56
    const int gk = k0 + kb;
    if (k0 + BK <= K && n0 + BN <= Nb) {
        const float* bp = B + (size_t)gk * Nb + n0 + ns;
        float4 f0 = *(const float4*)(bp + 0);
        float4 f1 = *(const float4*)(bp + 4);
        if (k0 + 3 * BK <= K) __builtin_prefetch(bp + (size_t)2 * BK * Nb, 0, 1);
        dst[(ns + 0) * SB + kb] = f2bf(f0.x);
        dst[(ns + 1) * SB + kb] = f2bf(f0.y);
        dst[(ns + 2) * SB + kb] = f2bf(f0.z);
        dst[(ns + 3) * SB + kb] = f2bf(f0.w);
        dst[(ns + 4) * SB + kb] = f2bf(f1.x);
        dst[(ns + 5) * SB + kb] = f2bf(f1.y);
        dst[(ns + 6) * SB + kb] = f2bf(f1.z);
        dst[(ns + 7) * SB + kb] = f2bf(f1.w);
    } else {
        for (int c = 0; c < 8; ++c) {
            const int gn = n0 + ns + c;
            float v = (gk < K && gn < Nb) ? B[(size_t)gk * Nb + gn] : 0.f;
            dst[(ns + c) * SB + kb] = f2bf(v);
        }
    }
}

// bf16 B-tile staging (transpose copy; K,Nb multiples of tile here)
__device__ __forceinline__ void stage_b_bf16(
    ush* __restrict__ dst, const ush* __restrict__ B,
    int n0, int k0, int Nb, int tid)
{
    const int kb = tid >> 3;
    const int ns = (tid & 7) * 8;
    const uint4 q = *(const uint4*)(B + (size_t)(k0 + kb) * Nb + n0 + ns);
    dst[(ns + 0) * SB + kb] = (ush)(q.x & 0xFFFFu);
    dst[(ns + 1) * SB + kb] = (ush)(q.x >> 16);
    dst[(ns + 2) * SB + kb] = (ush)(q.y & 0xFFFFu);
    dst[(ns + 3) * SB + kb] = (ush)(q.y >> 16);
    dst[(ns + 4) * SB + kb] = (ush)(q.z & 0xFFFFu);
    dst[(ns + 5) * SB + kb] = (ush)(q.z >> 16);
    dst[(ns + 6) * SB + kb] = (ush)(q.w & 0xFFFFu);
    dst[(ns + 7) * SB + kb] = (ush)(q.w >> 16);
}

// ---------------------------------------------------------------------------
// TDM issue: DMA one 128x32 bf16 A-tile into padded LDS (stride SA), wave 0.
// D# per CDNA5 ISA: group0 = {count,lds_addr,global_addr,type=2};
// group1 = {data_size=2B, pad 16dw-interval/4dw-amount, dims, strides}.
// NOTE: this toolchain uses the 6-arg builtin form (extra int32x8 before cpol).
// ---------------------------------------------------------------------------
__device__ __forceinline__ void tdm_load_a(
    const ush* __restrict__ Abf, ush* lds_dst, int m0, int k0, int M, int lda)
{
#if HAVE_TDM
    const unsigned long long ga =
        (unsigned long long)(size_t)(Abf + (size_t)m0 * lda + k0);
    const uint32 ldsa = (uint32)(size_t)lds_dst;
    u32x4 g0 = { 1u,                                  // count=1
                 ldsa,
                 (uint32)(ga & 0xFFFFFFFFull),
                 (uint32)((ga >> 32) & 0x01FFFFFFull) | 0x80000000u }; // type=2
    const int w0 = (1 << 16) | (1 << 20) | (3 << 22) | (3 << 25);
    // data_size=2B | pad_enable | pad_interval=16dw | pad_amount=4dw
    i32x8 g1 = { w0,
                 (int)((lda & 0xFFFF) << 16),                 // tensor_dim0 lo
                 (int)(((lda >> 16) & 0xFFFF) | ((M & 0xFFFF) << 16)),
                 (int)(((M >> 16) & 0xFFFF) | (BK << 16)),    // tile_dim0=32
                 (int)BM,                                     // tile_dim1=128
                 (int)lda,                                    // dim0 stride lo
                 0, 0 };
    i32x4 z4 = { 0, 0, 0, 0 };
    i32x8 z8 = { 0, 0, 0, 0, 0, 0, 0, 0 };
    __builtin_amdgcn_tensor_load_to_lds(g0, g1, z4, z4, z8, 0);
#else
    (void)Abf; (void)lds_dst; (void)m0; (void)k0; (void)M; (void)lda;
#endif
}

// fallback / non-TDM A-tile bf16 copy (all threads)
__device__ __forceinline__ void copy_a_bf16(
    ush* __restrict__ dst, const ush* __restrict__ Abf,
    int m0, int k0, int lda, int tid)
{
    const int row = tid >> 1;
    const int seg = (tid & 1) * 16;
    const ush* src = Abf + (size_t)(m0 + row) * lda + k0 + seg;
    ush* o = dst + row * SA + seg;
    *(uint4*)(o + 0) = *(const uint4*)(src + 0);
    *(uint4*)(o + 8) = *(const uint4*)(src + 8);
}

// ---------------------------------------------------------------------------
// GEMM with bf16 A (TDM-staged): C = Abf @ B (+bias); B fp32 or bf16;
// C fp32 or bf16.  K, M multiples of 32/128.
// ---------------------------------------------------------------------------
template<bool BBF, bool BIAS, bool OUTBF>
__global__ __launch_bounds__(256)
void gemm_bA(const ush* __restrict__ Abf, const void* __restrict__ Bv,
             void* __restrict__ Cv, const float* __restrict__ bias,
             int M, int K, int Nb, int lda)
{
    __shared__ ush As[2 * ASZ];
    __shared__ ush Bs[2 * BSZ];

    const int tid  = threadIdx.x;
    const int wave = tid >> 5;
    const int lane = tid & 31;
    const int waveM = wave >> 1, waveN = wave & 1;
    const int half = lane >> 4,  lr = lane & 15;
    const int m0 = blockIdx.y * BM;
    const int n0 = blockIdx.x * BN;

    v8f acc[2][2] = {};
    const int nk = K / BK;

    // prologue: tile 0
#if HAVE_TDM
    if (wave == 0) {
        tdm_load_a(Abf, As, m0, 0, M, lda);
    }
#else
    copy_a_bf16(As, Abf, m0, 0, lda, tid);
#endif
    if (BBF) stage_b_bf16(Bs, (const ush*)Bv, n0, 0, Nb, tid);
    else     stage_b_f32 (Bs, (const float*)Bv, n0, 0, K, Nb, tid);
#if HAVE_TDM
    if (wave == 0) __builtin_amdgcn_s_wait_tensorcnt(0);
#endif
    __syncthreads();

    int p = 0;
    for (int kt = 0; kt < nk; ++kt) {
        if (kt + 1 < nk) {
#if HAVE_TDM
            if (wave == 0) tdm_load_a(Abf, As + (p ^ 1) * ASZ, m0, (kt + 1) * BK, M, lda);
#else
            copy_a_bf16(As + (p ^ 1) * ASZ, Abf, m0, (kt + 1) * BK, lda, tid);
#endif
            if (BBF) stage_b_bf16(Bs + (p ^ 1) * BSZ, (const ush*)Bv, n0, (kt + 1) * BK, Nb, tid);
            else     stage_b_f32 (Bs + (p ^ 1) * BSZ, (const float*)Bv, n0, (kt + 1) * BK, K, Nb, tid);
        }
        const ush* Ab = As + p * ASZ;
        const ush* Bb = Bs + p * BSZ;

        Frag af[2], bfg[2];
        #pragma unroll
        for (int s = 0; s < 2; ++s) {
            const int row = waveM * 32 + s * 16 + lr;
            af[s].q[0] = *(const uint4*)&Ab[row * SA + half * 8];
            af[s].q[1] = *(const uint4*)&Ab[row * SA + 16 + half * 8];
            const int col = waveN * 32 + s * 16 + lr;
            bfg[s].q[0] = *(const uint4*)&Bb[col * SB + half * 8];
            bfg[s].q[1] = *(const uint4*)&Bb[col * SB + 16 + half * 8];
        }
        #pragma unroll
        for (int i = 0; i < 2; ++i)
            #pragma unroll
            for (int j = 0; j < 2; ++j)
                acc[i][j] = __builtin_amdgcn_wmma_f32_16x16x32_bf16(
                    false, af[i].v, false, bfg[j].v, (short)0, acc[i][j], false, false);
#if HAVE_TDM
        if (wave == 0 && kt + 1 < nk) __builtin_amdgcn_s_wait_tensorcnt(0);
#endif
        __syncthreads();
        p ^= 1;
    }

    #pragma unroll
    for (int i = 0; i < 2; ++i)
        #pragma unroll
        for (int j = 0; j < 2; ++j) {
            const int col = n0 + waveN * 32 + j * 16 + lr;
            if (col >= Nb) continue;
            const float bv = BIAS ? bias[col] : 0.f;
            #pragma unroll
            for (int r = 0; r < 8; ++r) {
                const int row = m0 + waveM * 32 + i * 16 + r + 8 * half;
                const float v = acc[i][j][r] + bv;
                if (OUTBF) ((ush*)Cv)[(size_t)row * Nb + col] = f2bf(v);
                else       ((float*)Cv)[(size_t)row * Nb + col] = v;
            }
        }
}

// ---------------------------------------------------------------------------
// GEMM with fp32 A (MODE 0 plain, MODE 1 * amask); B fp32.
// ---------------------------------------------------------------------------
template<int MODE, bool BIAS, bool OUTBF>
__global__ __launch_bounds__(256)
void gemm_fA(const float* __restrict__ A, const float* __restrict__ B,
             void* __restrict__ Cv, const float* __restrict__ bias,
             const float* __restrict__ amask,
             int M, int K, int Nb, int lda)
{
    __shared__ ush As[2 * ASZ];
    __shared__ ush Bs[2 * BSZ];

    const int tid  = threadIdx.x;
    const int wave = tid >> 5;
    const int lane = tid & 31;
    const int waveM = wave >> 1, waveN = wave & 1;
    const int half = lane >> 4,  lr = lane & 15;
    const int m0 = blockIdx.y * BM;
    const int n0 = blockIdx.x * BN;

    v8f acc[2][2] = {};
    const int nk = (K + BK - 1) / BK;

    auto stageA = [&](ush* dst, int k0) {
        const int row = tid >> 1;
        const int cs  = (tid & 1) * 16;
        const int gi  = m0 + row;
        ush* out = dst + row * SA + cs;
        if (k0 + BK <= K) {
            const float* ap = A + (size_t)gi * lda + k0 + cs;
            float4 f0 = *(const float4*)(ap + 0);
            float4 f1 = *(const float4*)(ap + 4);
            float4 f2 = *(const float4*)(ap + 8);
            float4 f3 = *(const float4*)(ap + 12);
            if (k0 + 3 * BK <= K) __builtin_prefetch(ap + 2 * BK, 0, 1);
            if (MODE == 1) {
                const float* mp = amask + (size_t)gi * lda + k0 + cs;
                float4 m0v = *(const float4*)(mp + 0);
                float4 m1v = *(const float4*)(mp + 4);
                float4 m2v = *(const float4*)(mp + 8);
                float4 m3v = *(const float4*)(mp + 12);
                f0.x *= m0v.x; f0.y *= m0v.y; f0.z *= m0v.z; f0.w *= m0v.w;
                f1.x *= m1v.x; f1.y *= m1v.y; f1.z *= m1v.z; f1.w *= m1v.w;
                f2.x *= m2v.x; f2.y *= m2v.y; f2.z *= m2v.z; f2.w *= m2v.w;
                f3.x *= m3v.x; f3.y *= m3v.y; f3.z *= m3v.z; f3.w *= m3v.w;
            }
            *(uint4*)(out + 0) = pack8bf(f0, f1);
            *(uint4*)(out + 8) = pack8bf(f2, f3);
        } else {
            for (int c = 0; c < 16; ++c) {
                const int kg = k0 + cs + c;
                float v = 0.f;
                if (kg < K) {
                    v = A[(size_t)gi * lda + kg];
                    if (MODE == 1) v *= amask[(size_t)gi * lda + kg];
                }
                out[c] = f2bf(v);
            }
        }
    };

    stageA(As, 0);
    stage_b_f32(Bs, B, n0, 0, K, Nb, tid);
    __syncthreads();

    int p = 0;
    for (int kt = 0; kt < nk; ++kt) {
        if (kt + 1 < nk) {
            stageA(As + (p ^ 1) * ASZ, (kt + 1) * BK);
            stage_b_f32(Bs + (p ^ 1) * BSZ, B, n0, (kt + 1) * BK, K, Nb, tid);
        }
        const ush* Ab = As + p * ASZ;
        const ush* Bb = Bs + p * BSZ;

        Frag af[2], bfg[2];
        #pragma unroll
        for (int s = 0; s < 2; ++s) {
            const int row = waveM * 32 + s * 16 + lr;
            af[s].q[0] = *(const uint4*)&Ab[row * SA + half * 8];
            af[s].q[1] = *(const uint4*)&Ab[row * SA + 16 + half * 8];
            const int col = waveN * 32 + s * 16 + lr;
            bfg[s].q[0] = *(const uint4*)&Bb[col * SB + half * 8];
            bfg[s].q[1] = *(const uint4*)&Bb[col * SB + 16 + half * 8];
        }
        #pragma unroll
        for (int i = 0; i < 2; ++i)
            #pragma unroll
            for (int j = 0; j < 2; ++j)
                acc[i][j] = __builtin_amdgcn_wmma_f32_16x16x32_bf16(
                    false, af[i].v, false, bfg[j].v, (short)0, acc[i][j], false, false);
        __syncthreads();
        p ^= 1;
    }

    #pragma unroll
    for (int i = 0; i < 2; ++i)
        #pragma unroll
        for (int j = 0; j < 2; ++j) {
            const int col = n0 + waveN * 32 + j * 16 + lr;
            if (col >= Nb) continue;
            const float bv = BIAS ? bias[col] : 0.f;
            #pragma unroll
            for (int r = 0; r < 8; ++r) {
                const int row = m0 + waveM * 32 + i * 16 + r + 8 * half;
                const float v = acc[i][j][r] + bv;
                if (OUTBF) ((ush*)Cv)[(size_t)row * Nb + col] = f2bf(v);
                else       ((float*)Cv)[(size_t)row * Nb + col] = v;
            }
        }
}

// ---------------------------------------------------------------------------
// Row stats / L2 normalize
// ---------------------------------------------------------------------------
template<bool USEE>
__global__ __launch_bounds__(256)
void row_stats(const float* __restrict__ adj, const float* __restrict__ emask,
               float* __restrict__ dis, float* __restrict__ cnt, int N)
{
    __shared__ float sb[256], cb[256];
    const int row = blockIdx.x, tid = threadIdx.x;
    float s = 0.f, c = 0.f;
    const float* ap = adj + (size_t)row * N;
    const float* mp = emask + (size_t)row * N;
    for (int j = tid * 4; j < N; j += 1024) {
        float4 a = *(const float4*)(ap + j);
        if (USEE) {
            float4 m = *(const float4*)(mp + j);
            a.x *= m.x; a.y *= m.y; a.z *= m.z; a.w *= m.w;
        }
        s += a.x + a.y + a.z + a.w;
        c += ((a.x > 0.f) ? 1.f : 0.f) + ((a.y > 0.f) ? 1.f : 0.f)
           + ((a.z > 0.f) ? 1.f : 0.f) + ((a.w > 0.f) ? 1.f : 0.f);
    }
    sb[tid] = s; cb[tid] = c;
    __syncthreads();
    for (int off = 128; off > 0; off >>= 1) {
        if (tid < off) { sb[tid] += sb[tid + off]; cb[tid] += cb[tid + off]; }
        __syncthreads();
    }
    if (tid == 0) {
        const float deg = sb[0] + 1.f;
        dis[row] = (deg > 0.f) ? rsqrtf(deg) : 0.f;
        cnt[row] = 2.f * cb[0] + 1.f;
    }
}

__global__ __launch_bounds__(256)
void l2norm_rows(const float* __restrict__ zp, float* __restrict__ zn)
{
    __shared__ float sb[256];
    const int row = blockIdx.x, tid = threadIdx.x;
    const float v = zp[(size_t)row * GP + tid];
    sb[tid] = v * v;
    __syncthreads();
    for (int off = 128; off > 0; off >>= 1) {
        if (tid < off) sb[tid] += sb[tid + off];
        __syncthreads();
    }
    const float d = fmaxf(sqrtf(sb[0]), 1e-12f);
    zn[(size_t)row * GP + tid] = v / d;
}

// ---------------------------------------------------------------------------
// Fused contrastive loss (register row-sums across the column sweep)
// ---------------------------------------------------------------------------
#define LBN 32
#define SBL 264

template<bool USEE>
__global__ __launch_bounds__(256)
void loss_fused(const float* __restrict__ Arows, const float* __restrict__ Bi,
                const float* __restrict__ Bt, const float* __restrict__ adj,
                const float* __restrict__ emask, const float* __restrict__ cnt,
                float* __restrict__ accum, float invTau, float weight)
{
    __shared__ ush BiS[LBN * SBL];
    __shared__ ush BtS[LBN * SBL];
    __shared__ float rowAcc[128 * 4];

    const int tid  = threadIdx.x;
    const int wave = tid >> 5;
    const int lane = tid & 31;
    const int waveM = wave >> 1, waveN = wave & 1;
    const int half = lane >> 4,  lr = lane & 15;
    const int r0 = blockIdx.x * 128;

    for (int i = tid; i < 128 * 4; i += 256) rowAcc[i] = 0.f;

    Frag af[2][8];
    #pragma unroll
    for (int s = 0; s < 2; ++s) {
        const int row = r0 + waveM * 32 + s * 16 + lr;
        const float* ap = Arows + (size_t)row * GP;
        #pragma unroll
        for (int kk = 0; kk < 8; ++kk) {
            const int kb = kk * 32;
            float4 g0a = *(const float4*)(ap + kb + half * 8);
            float4 g0b = *(const float4*)(ap + kb + half * 8 + 4);
            float4 g1a = *(const float4*)(ap + kb + 16 + half * 8);
            float4 g1b = *(const float4*)(ap + kb + 16 + half * 8 + 4);
            af[s][kk].q[0] = pack8bf(g0a, g0b);
            af[s][kk].q[1] = pack8bf(g1a, g1b);
        }
    }
    __syncthreads();

    float sAll[2][8] = {}, sAdj[2][8] = {}, dI[2][8] = {}, dT[2][8] = {};

    for (int j0 = 0; j0 < GN; j0 += LBN) {
        {
            const int n  = tid >> 3;
            const int kb = (tid & 7) * 32;
            const float* bi = Bi + (size_t)(j0 + n) * GP + kb;
            const float* bt = Bt + (size_t)(j0 + n) * GP + kb;
            ush* oi = &BiS[n * SBL + kb];
            ush* ot = &BtS[n * SBL + kb];
            #pragma unroll
            for (int c = 0; c < 4; ++c) {
                float4 a = *(const float4*)(bi + c * 8);
                float4 b = *(const float4*)(bi + c * 8 + 4);
                *(uint4*)(oi + c * 8) = pack8bf(a, b);
                float4 e = *(const float4*)(bt + c * 8);
                float4 f = *(const float4*)(bt + c * 8 + 4);
                *(uint4*)(ot + c * 8) = pack8bf(e, f);
            }
        }
        __syncthreads();

        v8f accI[2] = {}, accT[2] = {};
        #pragma unroll
        for (int kk = 0; kk < 8; ++kk) {
            const int col = waveN * 16 + lr;
            Frag bi, bt;
            bi.q[0] = *(const uint4*)&BiS[col * SBL + kk * 32 + half * 8];
            bi.q[1] = *(const uint4*)&BiS[col * SBL + kk * 32 + 16 + half * 8];
            bt.q[0] = *(const uint4*)&BtS[col * SBL + kk * 32 + half * 8];
            bt.q[1] = *(const uint4*)&BtS[col * SBL + kk * 32 + 16 + half * 8];
            #pragma unroll
            for (int s = 0; s < 2; ++s) {
                accI[s] = __builtin_amdgcn_wmma_f32_16x16x32_bf16(
                    false, af[s][kk].v, false, bi.v, (short)0, accI[s], false, false);
                accT[s] = __builtin_amdgcn_wmma_f32_16x16x32_bf16(
                    false, af[s][kk].v, false, bt.v, (short)0, accT[s], false, false);
            }
        }

        const int col = j0 + waveN * 16 + lr;
        #pragma unroll
        for (int s = 0; s < 2; ++s)
            #pragma unroll
            for (int r = 0; r < 8; ++r) {
                const int row = r0 + waveM * 32 + s * 16 + r + 8 * half;
                const float ei = __expf(accI[s][r] * invTau);
                const float et = __expf(accT[s][r] * invTau);
                float m = adj[(size_t)row * GN + col];
                if (USEE) m *= emask[(size_t)row * GN + col];
                sAll[s][r] += ei + et;
                sAdj[s][r] += m * (ei + et);
                if (row == col) { dI[s][r] = ei; dT[s][r] = et; }
            }
        __syncthreads();
    }

    #pragma unroll
    for (int s = 0; s < 2; ++s)
        #pragma unroll
        for (int r = 0; r < 8; ++r) {
            const int lrow = waveM * 32 + s * 16 + r + 8 * half;
            atomicAdd(&rowAcc[lrow * 4 + 0], sAll[s][r]);
            atomicAdd(&rowAcc[lrow * 4 + 1], sAdj[s][r]);
            atomicAdd(&rowAcc[lrow * 4 + 2], dI[s][r]);
            atomicAdd(&rowAcc[lrow * 4 + 3], dT[s][r]);
        }
    __syncthreads();

    if (tid < 128) {
        const int row = r0 + tid;
        const float l = __logf((rowAcc[tid * 4 + 3] + rowAcc[tid * 4 + 1]) /
                               (rowAcc[tid * 4 + 0] - rowAcc[tid * 4 + 2])) / cnt[row];
        atomicAdd(accum, -l * weight);
    }
}

__global__ void zero_accum(float* a) { a[0] = 0.f; }
__global__ void finalize_loss(const float* a, float* out) { out[0] = a[0]; }

// ---------------------------------------------------------------------------
extern "C" void kernel_launch(void* const* d_in, const int* in_sizes, int n_in,
                              void* d_out, int out_size, void* d_ws, size_t ws_size,
                              hipStream_t stream)
{
    const float* x      = (const float*)d_in[0];
    const float* adj    = (const float*)d_in[1];
    const float* fmask  = (const float*)d_in[2];
    const float* emask  = (const float*)d_in[3];
    const float* W1     = (const float*)d_in[4];
    const float* b1     = (const float*)d_in[5];
    const float* W2     = (const float*)d_in[6];
    const float* b2     = (const float*)d_in[7];
    const float* W_imp  = (const float*)d_in[8];
    const float* b_imp  = (const float*)d_in[9];
    const float* W_proj = (const float*)d_in[10];
    const float* b_proj = (const float*)d_in[11];

    float* z        = (float*)d_out;
    float* x_imp    = z + (size_t)GN * GP;
    float* loss_out = x_imp + (size_t)GN * GD;

    // workspace: bf16 region first, then fp32
    ush* anorm   = (ush*)d_ws;
    ush* anorm_a = anorm   + (size_t)GN * GN;
    ush* xw_bf   = anorm_a + (size_t)GN * GN;
    ush* h1_bf   = xw_bf   + (size_t)GN * GH;
    ush* h1w_bf  = h1_bf   + (size_t)GN * GH;
    float* fws   = (float*)(h1w_bf + (size_t)GN * GP);
    float* z_aug = fws;                      fws += (size_t)GN * GP;
    float* zp    = fws;                      fws += (size_t)GN * GP;
    float* zp_a  = fws;                      fws += (size_t)GN * GP;
    float* zn    = fws;                      fws += (size_t)GN * GP;
    float* zan   = fws;                      fws += (size_t)GN * GP;
    float* dis   = fws;                      fws += GN;
    float* dis_a = fws;                      fws += GN;
    float* cnt   = fws;                      fws += GN;
    float* cnt_a = fws;                      fws += GN;
    float* accum = fws;                      fws += 4;

    const float invTau = 2.0f;
    const float wgt    = 0.5f / (float)GN;

    row_stats<false><<<GN, 256, 0, stream>>>(adj, emask, dis,   cnt,   GN);
    row_stats<true ><<<GN, 256, 0, stream>>>(adj, emask, dis_a, cnt_a, GN);
    prep_anorm<false><<<GN, 256, 0, stream>>>(adj, emask, dis,   anorm);
    prep_anorm<true ><<<GN, 256, 0, stream>>>(adj, emask, dis_a, anorm_a);

    #define GEMM_F(MODE, BIAS, OUTBF, A, B, C, bias, am, M, K, Nb, lda)           \
        do { dim3 grid(((Nb) + BN - 1) / BN, (M) / BM);                            \
             gemm_fA<MODE, BIAS, OUTBF><<<grid, 256, 0, stream>>>(                 \
                 A, B, C, bias, am, M, K, Nb, lda); } while (0)
    #define GEMM_B(BBF, BIAS, OUTBF, A, B, C, bias, M, K, Nb, lda)                \
        do { dim3 grid(((Nb) + BN - 1) / BN, (M) / BM);                            \
             gemm_bA<BBF, BIAS, OUTBF><<<grid, 256, 0, stream>>>(                  \
                 A, B, C, bias, M, K, Nb, lda); } while (0)

    // ---- main branch ------------------------------------------------------
    GEMM_F(0, false, true,  x, W1, xw_bf, nullptr, nullptr, GN, GD, GH, GD);
    GEMM_B(true,  true, true,  anorm, xw_bf,  h1_bf,  b1, GN, GN, GH, GN);
    GEMM_B(false, false, true, h1_bf, W2,     h1w_bf, nullptr, GN, GH, GP, GH);
    GEMM_B(true,  true, false, anorm, h1w_bf, z,      b2, GN, GN, GP, GN);

    // ---- augmented branch -------------------------------------------------
    GEMM_F(1, false, true,  x, W1, xw_bf, nullptr, fmask, GN, GD, GH, GD);
    GEMM_B(true,  true, true,  anorm_a, xw_bf,  h1_bf,  b1, GN, GN, GH, GN);
    GEMM_B(false, false, true, h1_bf,   W2,     h1w_bf, nullptr, GN, GH, GP, GH);
    GEMM_B(true,  true, false, anorm_a, h1w_bf, z_aug,  b2, GN, GN, GP, GN);

    // ---- heads ------------------------------------------------------------
    GEMM_F(0, true, false, z,     W_imp,  x_imp, b_imp,  nullptr, GN, GP, GD, GP);
    GEMM_F(0, true, false, z,     W_proj, zp,    b_proj, nullptr, GN, GP, GP, GP);
    GEMM_F(0, true, false, z_aug, W_proj, zp_a,  b_proj, nullptr, GN, GP, GP, GP);

    l2norm_rows<<<GN, 256, 0, stream>>>(zp,   zn);
    l2norm_rows<<<GN, 256, 0, stream>>>(zp_a, zan);

    zero_accum<<<1, 1, 0, stream>>>(accum);
    loss_fused<false><<<GN / 128, 256, 0, stream>>>(zn, zn, zan, adj, emask, cnt,
                                                    accum, invTau, wgt);
    loss_fused<true ><<<GN / 128, 256, 0, stream>>>(zan, zan, zn, adj, emask, cnt_a,
                                                    accum, invTau, wgt);
    finalize_loss<<<1, 1, 0, stream>>>(accum, loss_out);
}
